// SAMCell_84129819394136
// MI455X (gfx1250) — matrix operand order repaired and checked by reading, not compiled, not run on hardware
//
#include <hip/hip_runtime.h>
#include <hip/hip_bf16.h>

typedef __attribute__((ext_vector_type(16))) _Float16 v16h;
typedef __attribute__((ext_vector_type(8)))  _Float16 v8h;
typedef __attribute__((ext_vector_type(8)))  float    v8f;

#define MEM_DIM  256
#define EDGE_DIM 172
#define TIME_DIM 100
#define QIN      528
#define QIN_PAD  544   // 17 * 32
#define KSTEPS   17
#define NPROTO   5
#define NTILES   16    // 256 / 16 output column tiles

// packed-B layout: wB[(((ntile*KSTEPS)+ks)*32 + lane)*16 + e]
//   e -> K = ks*32 + e + (lane>=16 ? 16 : 0), N = ntile*16 + (lane&15)
#define PACKB_ELEMS ((size_t)NTILES * KSTEPS * 32 * 16)

__device__ __forceinline__ float nan_clean(float x) {
  if (__builtin_isnan(x)) return 0.f;
  if (__builtin_isinf(x)) return x > 0.f ? 10.f : -10.f;
  return x;
}
__device__ __forceinline__ float clampf(float x, float lo, float hi) {
  return fminf(fmaxf(x, lo), hi);
}
// full butterfly reduction within each contiguous 16-lane group (wave32)
__device__ __forceinline__ float reduce16(float v) {
  v += __shfl_xor(v, 1, 32);
  v += __shfl_xor(v, 2, 32);
  v += __shfl_xor(v, 4, 32);
  v += __shfl_xor(v, 8, 32);
  return v;
}
__device__ __forceinline__ v8f wmma_f16(v16h a, v16h b, v8f c) {
  return __builtin_amdgcn_wmma_f32_16x16x32_f16(false, a, false, b, (short)0, c,
                                                false, false);
}

// one-time repack of Wq (fp32, [256,528] row-major) into WMMA-B-native f16 frags
__global__ __launch_bounds__(256)
void pack_wq(const float* __restrict__ Wq, _Float16* __restrict__ wB) {
  const int idx = blockIdx.x * 256 + threadIdx.x;
  if (idx >= (int)PACKB_ELEMS) return;
  const int e     = idx & 15;
  const int lane  = (idx >> 4) & 31;
  const int blk   = idx >> 9;              // ntile*KSTEPS + ks
  const int ntile = blk / KSTEPS;
  const int ks    = blk - ntile * KSTEPS;
  const int K     = ks * 32 + e + ((lane >> 4) ? 16 : 0);
  const int N     = ntile * 16 + (lane & 15);
  const float v   = (K < QIN) ? Wq[(size_t)N * QIN + K] : 0.f;
  wB[idx] = (_Float16)v;
}

template <bool PACKED>
__global__ __launch_bounds__(256)
void samcell_fused(const float* __restrict__ raw,
                   const float* __restrict__ edgef,
                   const float* __restrict__ timeenc,
                   const float* __restrict__ proto,
                   const float* __restrict__ Wq,
                   const _Float16* __restrict__ wB,
                   const float* __restrict__ bq,
                   const float* __restrict__ Wg,
                   const float* __restrict__ bg,
                   const float* __restrict__ temperature,
                   const float* __restrict__ gamma_,
                   const float* __restrict__ beta_,
                   const int*   __restrict__ mask,
                   float* __restrict__ out,
                   int Bn)
{
  __shared__ _Float16 sQin[16 * QIN_PAD];   // 17408 B : f16 concat inputs, zero-padded
  __shared__ float    sQ[16 * MEM_DIM];     // 16384 B : pre-activation query tile

  const int tile    = blockIdx.x;
  const int rowBase = tile * 16;
  const int tid     = threadIdx.x;

  // ---- prefetch this tile's prototype block (phase-2's dominant HBM traffic) ----
  {
    const size_t pOff   = (size_t)rowBase * NPROTO * MEM_DIM;
    const size_t rows   = (size_t)((rowBase + 16 <= Bn) ? 16 : (Bn - rowBase));
    const size_t pbytes = rows * NPROTO * MEM_DIM * sizeof(float);   // up to 80 KB
    const char*  pbase  = (const char*)(proto + pOff);
    for (size_t off = (size_t)tid * 128; off < pbytes; off += 256 * 128)
      __builtin_prefetch(pbase + off, 0, 0);
  }

  // ---- stage nan-cleaned concat([raw,edge,time]) as f16 into LDS ----
  for (int idx = tid; idx < 16 * QIN_PAD; idx += 256) {
    const int r  = idx / QIN_PAD;
    const int k  = idx - r * QIN_PAD;
    const int gr = rowBase + r;
    float v = 0.f;
    if (gr < Bn) {
      if (k < MEM_DIM)                 v = nan_clean(raw[(size_t)gr * MEM_DIM + k]);
      else if (k < MEM_DIM + EDGE_DIM) v = nan_clean(edgef[(size_t)gr * EDGE_DIM + (k - MEM_DIM)]);
      else if (k < QIN)                v = timeenc[(size_t)gr * TIME_DIM + (k - (MEM_DIM + EDGE_DIM))];
    }
    sQin[idx] = (_Float16)v;
  }
  __syncthreads();

  // ---- WMMA GEMM: query_pre[16,256] = Qin[16,544] @ WqT[544,256] ----
  const int lane = tid & 31;
  const int wv   = tid >> 5;       // wave 0..7
  const int lh   = lane >> 4;      // lane half
  const int ll   = lane & 15;
  const int n0   = (wv * 2 + 0) * 16;
  const int n1   = (wv * 2 + 1) * 16;

  v8f acc0 = {};
  v8f acc1 = {};

  // A fragment loader: contiguous k order, same as packed B
  // (lane<16: K{0..15}; lane>=16: K{16..31})
  const int aBase = ll * QIN_PAD + lh * 16;

  if (PACKED) {
    const _Float16* pB0 = wB + ((size_t)(wv * 2 + 0) * KSTEPS * 32 + lane) * 16;
    const _Float16* pB1 = wB + ((size_t)(wv * 2 + 1) * KSTEPS * 32 + lane) * 16;
    #pragma unroll 2
    for (int ks = 0; ks < KSTEPS; ++ks) {
      v16h a;
      {
        const v8h lo = *(const v8h*)(&sQin[aBase + ks * 32]);
        const v8h hi = *(const v8h*)(&sQin[aBase + ks * 32 + 8]);
        #pragma unroll
        for (int e = 0; e < 8; ++e) { a[e] = lo[e]; a[e + 8] = hi[e]; }
      }
      const v16h b0 = *(const v16h*)(pB0 + (size_t)ks * 512);   // 32B aligned
      const v16h b1 = *(const v16h*)(pB1 + (size_t)ks * 512);
      acc0 = wmma_f16(a, b0, acc0);
      acc1 = wmma_f16(a, b1, acc1);
    }
  } else {
    // fallback: convert Wq f32->f16 on the fly; tail k-step peeled
    #pragma unroll 2
    for (int ks = 0; ks < KSTEPS - 1; ++ks) {
      v16h a;
      {
        const v8h lo = *(const v8h*)(&sQin[aBase + ks * 32]);
        const v8h hi = *(const v8h*)(&sQin[aBase + ks * 32 + 8]);
        #pragma unroll
        for (int e = 0; e < 8; ++e) { a[e] = lo[e]; a[e + 8] = hi[e]; }
      }
      const int kstart = ks * 32 + lh * 16;                      // always < 528-15
      const float* s0 = Wq + (size_t)(n0 + ll) * QIN + kstart;   // 64B aligned
      const float* s1 = Wq + (size_t)(n1 + ll) * QIN + kstart;
      v16h b0, b1;
      #pragma unroll
      for (int e = 0; e < 16; ++e) { b0[e] = (_Float16)s0[e]; b1[e] = (_Float16)s1[e]; }
      acc0 = wmma_f16(a, b0, acc0);
      acc1 = wmma_f16(a, b1, acc1);
    }
    {   // ks = 16 : lane-half 1 covers K 528..543 -> zeros
      const int ks = KSTEPS - 1;
      v16h a;
      {
        const v8h lo = *(const v8h*)(&sQin[aBase + ks * 32]);
        const v8h hi = *(const v8h*)(&sQin[aBase + ks * 32 + 8]);
        #pragma unroll
        for (int e = 0; e < 8; ++e) { a[e] = lo[e]; a[e + 8] = hi[e]; }
      }
      v16h b0, b1;
      #pragma unroll
      for (int e = 0; e < 16; ++e) { b0[e] = (_Float16)0.f; b1[e] = (_Float16)0.f; }
      if (!lh) {
        const int kstart = ks * 32;
        const float* s0 = Wq + (size_t)(n0 + ll) * QIN + kstart;
        const float* s1 = Wq + (size_t)(n1 + ll) * QIN + kstart;
        #pragma unroll
        for (int e = 0; e < 16; ++e) { b0[e] = (_Float16)s0[e]; b1[e] = (_Float16)s1[e]; }
      }
      acc0 = wmma_f16(a, b0, acc0);
      acc1 = wmma_f16(a, b1, acc1);
    }
  }

  // epilogue: C/D layout elem j -> (M = j + 8*lh, N = ll); add bq, spill tile to LDS
  {
    const float bq0 = bq[n0 + ll];
    const float bq1 = bq[n1 + ll];
    #pragma unroll
    for (int j = 0; j < 8; ++j) {
      sQ[(j + 8 * lh) * MEM_DIM + n0 + ll] = acc0[j] + bq0;
      sQ[(j + 8 * lh) * MEM_DIM + n1 + ll] = acc1[j] + bq1;
    }
  }
  __syncthreads();

  // ---- phase 2: 16 threads per row, 16 cols per thread ----
  const int row = tid >> 4;
  const int c   = tid & 15;
  const int gr  = rowBase + row;
  if (gr >= Bn) return;            // uniform per 16-lane group
  const int colBase = c * 16;

  float* out_updated = out;
  float* out_attn    = out + (size_t)Bn * 256;
  float* out_gate    = out + (size_t)Bn * 261;
  float* out_cand    = out + (size_t)Bn * 262;
  float* out_query   = out + (size_t)Bn * 518;
  float* out_sim     = out + (size_t)Bn * 774;

  float qp[16];
  #pragma unroll
  for (int i = 0; i < 16; ++i) qp[i] = sQ[row * MEM_DIM + colBase + i];

  // layernorm + tanh
  float s = 0.f;
  #pragma unroll
  for (int i = 0; i < 16; ++i) s += qp[i];
  const float mean = reduce16(s) * (1.f / 256.f);
  float s2 = 0.f;
  #pragma unroll
  for (int i = 0; i < 16; ++i) { float d = qp[i] - mean; s2 += d * d; }
  const float rstd = rsqrtf(reduce16(s2) * (1.f / 256.f) + 1e-5f);

  float gm[16], bt[16], qv[16];
  #pragma unroll
  for (int i = 0; i < 16; ++i) { gm[i] = gamma_[colBase + i]; bt[i] = beta_[colBase + i]; }
  #pragma unroll
  for (int i = 0; i < 16; ++i) qv[i] = tanhf((qp[i] - mean) * rstd * gm[i] + bt[i]);
  #pragma unroll
  for (int i = 0; i < 16; ++i) out_query[(size_t)gr * 256 + colBase + i] = qv[i];

  // cosine sims vs prototypes
  float qsq = 0.f;
  #pragma unroll
  for (int i = 0; i < 16; ++i) qsq += qv[i] * qv[i];
  const float qn  = fmaxf(sqrtf(reduce16(qsq)), 1e-12f);
  const float tcl = clampf(temperature[0], 0.05f, 2.0f) + 1e-6f;
  const float* prRow = proto + (size_t)gr * NPROTO * 256;

  float simv[NPROTO];
  #pragma unroll
  for (int p = 0; p < NPROTO; ++p) {
    float dot = 0.f, psq = 0.f;
    #pragma unroll
    for (int i = 0; i < 16; ++i) {
      const float pv = prRow[p * 256 + colBase + i];
      dot += qv[i] * pv;
      psq += pv * pv;
    }
    dot = reduce16(dot);
    psq = reduce16(psq);
    const float pn = fmaxf(sqrtf(psq), 1e-12f);
    simv[p] = (dot / (qn * pn)) / tcl;
  }

  // masked softmax with uniform fallback
  int m[NPROTO]; bool any = false;
  #pragma unroll
  for (int p = 0; p < NPROTO; ++p) { m[p] = mask[(size_t)gr * NPROTO + p]; any = any || (m[p] != 0); }
  float sm[NPROTO], safe[NPROTO];
  #pragma unroll
  for (int p = 0; p < NPROTO; ++p) {
    sm[p]   = m[p] ? simv[p] : -__builtin_inff();
    safe[p] = any ? sm[p] : 0.f;
  }
  float mx = safe[0];
  #pragma unroll
  for (int p = 1; p < NPROTO; ++p) mx = fmaxf(mx, safe[p]);
  float es = 0.f, attn[NPROTO];
  #pragma unroll
  for (int p = 0; p < NPROTO; ++p) { attn[p] = expf(safe[p] - mx); es += attn[p]; }
  const float inv_es = 1.f / es;
  #pragma unroll
  for (int p = 0; p < NPROTO; ++p) attn[p] *= inv_es;

  // candidate
  float cn[16];
  #pragma unroll
  for (int i = 0; i < 16; ++i) cn[i] = 0.f;
  #pragma unroll
  for (int p = 0; p < NPROTO; ++p) {
    #pragma unroll
    for (int i = 0; i < 16; ++i) cn[i] += attn[p] * prRow[p * 256 + colBase + i];
  }
  #pragma unroll
  for (int i = 0; i < 16; ++i) cn[i] = clampf(cn[i], -5.f, 5.f);
  #pragma unroll
  for (int i = 0; i < 16; ++i) out_cand[(size_t)gr * 256 + colBase + i] = cn[i];

  // gate: sigmoid(dot(clip(concat[raw,cand,time]), Wg) + bg)
  float rw[16];
  #pragma unroll
  for (int i = 0; i < 16; ++i) rw[i] = nan_clean(raw[(size_t)gr * 256 + colBase + i]);

  float gd = 0.f;
  #pragma unroll
  for (int i = 0; i < 16; ++i) gd += clampf(rw[i], -100.f, 100.f) * Wg[colBase + i];
  #pragma unroll
  for (int i = 0; i < 16; ++i) gd += clampf(cn[i], -100.f, 100.f) * Wg[256 + colBase + i];
  for (int t = c; t < TIME_DIM; t += 16)
    gd += clampf(timeenc[(size_t)gr * TIME_DIM + t], -100.f, 100.f) * Wg[512 + t];
  gd = reduce16(gd);
  const float gate = 1.f / (1.f + expf(-(gd + bg[0])));

  if (c == 0) {
    out_gate[gr] = gate;
    #pragma unroll
    for (int p = 0; p < NPROTO; ++p) {
      out_attn[(size_t)gr * NPROTO + p] = attn[p];
      out_sim [(size_t)gr * NPROTO + p] = sm[p];
    }
  }

  // gated blend + layernorm + clip
  float up[16];
  #pragma unroll
  for (int i = 0; i < 16; ++i) up[i] = (1.f - gate) * rw[i] + gate * cn[i];
  float us = 0.f;
  #pragma unroll
  for (int i = 0; i < 16; ++i) us += up[i];
  const float um = reduce16(us) * (1.f / 256.f);
  float uvar = 0.f;
  #pragma unroll
  for (int i = 0; i < 16; ++i) { float d = up[i] - um; uvar += d * d; }
  const float urs = rsqrtf(reduce16(uvar) * (1.f / 256.f) + 1e-5f);
  #pragma unroll
  for (int i = 0; i < 16; ++i)
    out_updated[(size_t)gr * 256 + colBase + i] =
        clampf((up[i] - um) * urs * gm[i] + bt[i], -50.f, 50.f);
}

extern "C" void kernel_launch(void* const* d_in, const int* in_sizes, int n_in,
                              void* d_out, int out_size, void* d_ws, size_t ws_size,
                              hipStream_t stream) {
  const float* raw      = (const float*)d_in[0];
  // d_in[1] = node_features (unused by reference)
  const float* edgef    = (const float*)d_in[2];
  const float* timeenc  = (const float*)d_in[3];
  const float* proto    = (const float*)d_in[4];
  const float* Wq       = (const float*)d_in[5];
  const float* bq       = (const float*)d_in[6];
  const float* Wg       = (const float*)d_in[7];
  const float* bg       = (const float*)d_in[8];
  const float* temp     = (const float*)d_in[9];
  const float* gamma_   = (const float*)d_in[10];
  const float* beta_    = (const float*)d_in[11];
  const int*   maskp    = (const int*)d_in[12];
  float* out = (float*)d_out;

  const int Bn    = in_sizes[0] / MEM_DIM;
  const int tiles = (Bn + 15) / 16;

  const size_t packBytes = PACKB_ELEMS * sizeof(_Float16);   // 278,528 B
  const bool usePacked = (d_ws != nullptr) && (ws_size >= packBytes);

  if (usePacked) {
    _Float16* wB = (_Float16*)d_ws;
    const int packBlocks = (int)((PACKB_ELEMS + 255) / 256);
    pack_wq<<<packBlocks, 256, 0, stream>>>(Wq, wB);
    samcell_fused<true><<<tiles, 256, 0, stream>>>(raw, edgef, timeenc, proto,
                                                   Wq, wB, bq, Wg, bg, temp,
                                                   gamma_, beta_, maskp, out, Bn);
  } else {
    samcell_fused<false><<<tiles, 256, 0, stream>>>(raw, edgef, timeenc, proto,
                                                    Wq, nullptr, bq, Wg, bg, temp,
                                                    gamma_, beta_, maskp, out, Bn);
  }
}
